// MultiHeadAttention_8675833938350
// MI455X (gfx1250) — compile-verified, matrix operand order checked by reading, not run on hardware
//
#include <hip/hip_runtime.h>

// ---------- types ----------
typedef __attribute__((ext_vector_type(16))) __bf16 v16bf;
typedef __attribute__((ext_vector_type(8)))  __bf16 bf16x8;
typedef __attribute__((ext_vector_type(8)))  float  v8f;
typedef __attribute__((ext_vector_type(4)))  float  f32x4;

static constexpr int B_  = 2;
static constexpr int S_  = 2048;
static constexpr int D_  = 1024;
static constexpr int H_  = 16;
static constexpr int DK_ = 64;
static constexpr int STR = 2052;           // LDS score-row stride (floats), padded

// ---------- fragment helpers ----------
// 16-bit A/B fragment (16x32 / 32x16): lane holds two contiguous 8-elem runs
// at k0 = 8*(lane>>4) and k0+16 (per ISA 7.12.2 tables).
__device__ __forceinline__ v16bf load_frag_bf16(const __bf16* p, int k0) {
    bf16x8 lo = *(const bf16x8*)(p + k0);
    bf16x8 hi = *(const bf16x8*)(p + k0 + 16);
    v16bf r;
#pragma unroll
    for (int i = 0; i < 8; ++i) { r[i] = lo[i]; r[i + 8] = hi[i]; }
    return r;
}

__device__ __forceinline__ v16bf load_frag_f32(const float* p, int k0) {
    f32x4 a0 = *(const f32x4*)(p + k0);
    f32x4 a1 = *(const f32x4*)(p + k0 + 4);
    f32x4 b0 = *(const f32x4*)(p + k0 + 16);
    f32x4 b1 = *(const f32x4*)(p + k0 + 20);
    v16bf r;
#pragma unroll
    for (int i = 0; i < 4; ++i) {
        r[i]      = (__bf16)a0[i];
        r[i + 4]  = (__bf16)a1[i];
        r[i + 8]  = (__bf16)b0[i];
        r[i + 12] = (__bf16)b1[i];
    }
    return r;
}

__device__ __forceinline__ v8f wmma_bf16(v16bf a, v16bf b, v8f c) {
    return __builtin_amdgcn_wmma_f32_16x16x32_bf16(
        /*neg_a=*/false, a, /*neg_b=*/false, b,
        /*c_mod=*/(short)0, c, /*reuse_a=*/false, /*reuse_b=*/false);
}

// ---------- fp32 -> bf16 weight conversion ----------
__global__ void cvt_kernel(const float* __restrict__ src, __bf16* __restrict__ dst, int n) {
    int i = blockIdx.x * blockDim.x + threadIdx.x;
    if (i < n) dst[i] = (__bf16)src[i];
}

// ---------- projection GEMM: Y = X @ W^T + b ----------
// X: [4096, 1024] fp32 ; W: [1024,1024] bf16 (row-major, row n = output col n)
// vmode 0: dst[b,h,s,dk] (Q,K layout)     vmode 1: dst[b,h,dk,s] (V transposed)
__global__ void __launch_bounds__(256, 1)
proj_gemm_kernel(const float* __restrict__ X,
                 const __bf16* __restrict__ Wb,
                 const float* __restrict__ bias,
                 __bf16* __restrict__ dst,
                 int vmode) {
    const int lane  = threadIdx.x & 31;
    const int wave  = threadIdx.x >> 5;
    const int gwt   = blockIdx.x * 8 + wave;     // 4096 wave tiles
    const int mbase = (gwt >> 4) * 16;           // 256 m-tiles
    const int nbase = (gwt & 15) * 64;           // 16 n-blocks of 64
    const int l15   = lane & 15;
    const int khalf = lane >> 4;
    const int k0    = khalf * 8;

    const float*  xrow = X + (size_t)(mbase + l15) * D_;
    const __bf16* wrow[4];
#pragma unroll
    for (int j = 0; j < 4; ++j)
        wrow[j] = Wb + (size_t)(nbase + j * 16 + l15) * D_;

    v8f acc[4] = {{}, {}, {}, {}};

    // software pipeline: frags for k are loaded while WMMAs for k-32 execute
    v16bf a_c = load_frag_f32(xrow, k0);
    v16bf b_c[4];
#pragma unroll
    for (int j = 0; j < 4; ++j) b_c[j] = load_frag_bf16(wrow[j], k0);

#pragma unroll 2
    for (int k = 32; k < D_; k += 32) {
        v16bf a_n = load_frag_f32(xrow + k, k0);
        v16bf b_n[4];
#pragma unroll
        for (int j = 0; j < 4; ++j) b_n[j] = load_frag_bf16(wrow[j] + k, k0);
#pragma unroll
        for (int j = 0; j < 4; ++j) acc[j] = wmma_bf16(a_c, b_c[j], acc[j]);
        a_c = a_n;
#pragma unroll
        for (int j = 0; j < 4; ++j) b_c[j] = b_n[j];
    }
#pragma unroll
    for (int j = 0; j < 4; ++j) acc[j] = wmma_bf16(a_c, b_c[j], acc[j]);

#pragma unroll
    for (int j = 0; j < 4; ++j) {
        const int   n  = nbase + j * 16 + l15;
        const float bn = bias[n];
        const int   h  = n >> 6;
        const int   dk = n & 63;
#pragma unroll
        for (int r = 0; r < 8; ++r) {
            const int m  = mbase + khalf * 8 + r;   // C layout: row = r + 8*(lane>>4)
            const int bb = m >> 11;                 // m / S_
            const int s  = m & (S_ - 1);
            const float v = acc[j][r] + bn;
            size_t addr;
            if (vmode == 0)
                addr = (((size_t)(bb * H_ + h)) * S_ + s) * DK_ + dk;
            else
                addr = (((size_t)(bb * H_ + h)) * DK_ + dk) * (size_t)S_ + s;
            dst[addr] = (__bf16)v;
        }
    }
}

// ---------- fused attention: scores -> softmax -> PV (per 16-row query tile) ----------
__global__ void __launch_bounds__(256, 1)
attn_kernel(const __bf16* __restrict__ Q,
            const __bf16* __restrict__ K,
            const __bf16* __restrict__ Vt,
            float* __restrict__ attnW,
            __bf16* __restrict__ ctx) {
    extern __shared__ float smem[];                 // 16*STR score tile + 256 reduce
    const int id    = blockIdx.x;                   // B*H*(S/16) = 4096 blocks
    const int mt    = id & 127;
    const int bh    = id >> 7;                      // b*H + h
    const int h     = bh & (H_ - 1);
    const int b     = bh >> 4;
    const int mbase = mt * 16;
    const int tid   = threadIdx.x;
    const int lane  = tid & 31;
    const int wave  = tid >> 5;
    const int l15   = lane & 15;
    const int khalf = lane >> 4;
    const int k0    = khalf * 8;
    const size_t headQK = (size_t)bh * S_ * DK_;
    const int cwave = wave * 256;                   // this wave's key-column range

    // ---- Phase 1: score tile 16 x 2048 into LDS ----
    const __bf16* qrow  = Q + headQK + (size_t)(mbase + l15) * DK_;
    const __bf16* kbase = K + headQK;
    v16bf a0 = load_frag_bf16(qrow, k0);
    v16bf a1 = load_frag_bf16(qrow + 32, k0);

    const __bf16* kr0 = kbase + (size_t)(cwave + l15) * DK_;
    v16bf bc0 = load_frag_bf16(kr0, k0);
    v16bf bc1 = load_frag_bf16(kr0 + 32, k0);
#pragma unroll 2
    for (int nf = 0; nf < 16; ++nf) {
        v16bf bn0 = bc0, bn1 = bc1;
        if (nf < 15) {
            const __bf16* nk = kbase + (size_t)(cwave + (nf + 1) * 16 + l15) * DK_;
            bn0 = load_frag_bf16(nk, k0);
            bn1 = load_frag_bf16(nk + 32, k0);
        }
        v8f acc = {};
        acc = wmma_bf16(a0, bc0, acc);
        acc = wmma_bf16(a1, bc1, acc);
        const int c = cwave + nf * 16;
#pragma unroll
        for (int r = 0; r < 8; ++r) {
            const int row = khalf * 8 + r;
            smem[row * STR + c + l15] = acc[r] * 0.125f;   // 1/sqrt(64)
        }
        bc0 = bn0; bc1 = bn1;
    }
    __syncthreads();

    // ---- Phase 2: row softmax (16 rows x 16 threads each) + stream P out ----
    {
        float* red = smem + 16 * STR;
        const int row = tid >> 4;
        const int sub = tid & 15;
        float* srow = smem + row * STR;
        float mx = -3.4e38f;
        for (int i = 0; i < 128; ++i) mx = fmaxf(mx, srow[sub + 16 * i]);
        red[tid] = mx; __syncthreads();
        for (int off = 8; off >= 1; off >>= 1) {
            if (sub < off) red[tid] = fmaxf(red[tid], red[tid + off]);
            __syncthreads();
        }
        const float rowmax = red[row * 16]; __syncthreads();
        float ssum = 0.f;
        for (int i = 0; i < 128; ++i) {
            const int c = sub + 16 * i;
            const float e = __expf(srow[c] - rowmax);
            srow[c] = e; ssum += e;
        }
        red[tid] = ssum; __syncthreads();
        for (int off = 8; off >= 1; off >>= 1) {
            if (sub < off) red[tid] += red[tid + off];
            __syncthreads();
        }
        const float inv = 1.f / red[row * 16]; __syncthreads();
        float* wout = attnW + ((size_t)bh * S_ + mbase + row) * (size_t)S_;
        for (int i = 0; i < 128; ++i) {
            const int c = sub + 16 * i;
            const float p = srow[c] * inv;
            srow[c] = p;
            wout[c] = p;                            // required output, coalesced
        }
    }
    __syncthreads();

    // ---- Phase 3: context partial = P(16 x 256) @ V(256 x 64), per wave ----
    v8f acc[4] = {{}, {}, {}, {}};
    const __bf16* vbase = Vt + (size_t)bh * DK_ * S_;
    const float*  prow  = smem + (size_t)l15 * STR + cwave;
    const __bf16* vrow[4];
#pragma unroll
    for (int j = 0; j < 4; ++j)
        vrow[j] = vbase + (size_t)(j * 16 + l15) * S_ + cwave;

    v16bf ap_c = load_frag_f32(prow, k0);           // P fp32 -> bf16 from LDS
    v16bf vc[4];
#pragma unroll
    for (int j = 0; j < 4; ++j) vc[j] = load_frag_bf16(vrow[j], k0);

#pragma unroll 2
    for (int kk = 32; kk < 256; kk += 32) {
        v16bf ap_n = load_frag_f32(prow + kk, k0);
        v16bf vn[4];
#pragma unroll
        for (int j = 0; j < 4; ++j) vn[j] = load_frag_bf16(vrow[j] + kk, k0);
#pragma unroll
        for (int j = 0; j < 4; ++j) acc[j] = wmma_bf16(ap_c, vc[j], acc[j]);
        ap_c = ap_n;
#pragma unroll
        for (int j = 0; j < 4; ++j) vc[j] = vn[j];
    }
#pragma unroll
    for (int j = 0; j < 4; ++j) acc[j] = wmma_bf16(ap_c, vc[j], acc[j]);
    __syncthreads();                                // all waves done reading P

    // write wave partials (reuse score LDS region: 8 x 1024 fp32 = 32KB)
#pragma unroll
    for (int j = 0; j < 4; ++j) {
        const int dk = j * 16 + l15;
#pragma unroll
        for (int r = 0; r < 8; ++r) {
            const int row = khalf * 8 + r;
            smem[wave * 1024 + row * 64 + dk] = acc[j][r];
        }
    }
    __syncthreads();

    // cross-wave reduce + store context bf16 merged-head layout [B,S,D]
    for (int e = tid; e < 1024; e += 256) {
        float s = 0.f;
#pragma unroll
        for (int w = 0; w < 8; ++w) s += smem[w * 1024 + e];
        const int row = e >> 6, dk = e & 63;
        ctx[((size_t)(b * S_ + mbase + row)) * D_ + h * DK_ + dk] = (__bf16)s;
    }
}

// ---------- output GEMM: out = ctx @ W_o^T + b_o (fp32 out) ----------
__global__ void __launch_bounds__(256, 1)
out_gemm_kernel(const __bf16* __restrict__ A,
                const __bf16* __restrict__ Wb,
                const float* __restrict__ bias,
                float* __restrict__ out) {
    const int lane  = threadIdx.x & 31;
    const int wave  = threadIdx.x >> 5;
    const int gwt   = blockIdx.x * 8 + wave;
    const int mbase = (gwt >> 4) * 16;
    const int nbase = (gwt & 15) * 64;
    const int l15   = lane & 15;
    const int khalf = lane >> 4;
    const int k0    = khalf * 8;

    const __bf16* arow = A + (size_t)(mbase + l15) * D_;
    const __bf16* wrow[4];
#pragma unroll
    for (int j = 0; j < 4; ++j)
        wrow[j] = Wb + (size_t)(nbase + j * 16 + l15) * D_;

    v8f acc[4] = {{}, {}, {}, {}};

    v16bf a_c = load_frag_bf16(arow, k0);
    v16bf b_c[4];
#pragma unroll
    for (int j = 0; j < 4; ++j) b_c[j] = load_frag_bf16(wrow[j], k0);

#pragma unroll 2
    for (int k = 32; k < D_; k += 32) {
        v16bf a_n = load_frag_bf16(arow + k, k0);
        v16bf b_n[4];
#pragma unroll
        for (int j = 0; j < 4; ++j) b_n[j] = load_frag_bf16(wrow[j] + k, k0);
#pragma unroll
        for (int j = 0; j < 4; ++j) acc[j] = wmma_bf16(a_c, b_c[j], acc[j]);
        a_c = a_n;
#pragma unroll
        for (int j = 0; j < 4; ++j) b_c[j] = b_n[j];
    }
#pragma unroll
    for (int j = 0; j < 4; ++j) acc[j] = wmma_bf16(a_c, b_c[j], acc[j]);

#pragma unroll
    for (int j = 0; j < 4; ++j) {
        const int   n  = nbase + j * 16 + l15;
        const float bn = bias[n];
#pragma unroll
        for (int r = 0; r < 8; ++r) {
            const int m = mbase + khalf * 8 + r;
            out[(size_t)m * D_ + n] = acc[j][r] + bn;
        }
    }
}

// ---------- host launcher ----------
extern "C" void kernel_launch(void* const* d_in, const int* in_sizes, int n_in,
                              void* d_out, int out_size, void* d_ws, size_t ws_size,
                              hipStream_t stream) {
    (void)in_sizes; (void)n_in; (void)out_size; (void)ws_size;
    const float* q  = (const float*)d_in[0];
    const float* k  = (const float*)d_in[1];
    const float* v  = (const float*)d_in[2];
    const float* Wq = (const float*)d_in[3];
    const float* bq = (const float*)d_in[4];
    const float* Wk = (const float*)d_in[5];
    const float* bk = (const float*)d_in[6];
    const float* Wv = (const float*)d_in[7];
    const float* bv = (const float*)d_in[8];
    const float* Wo = (const float*)d_in[9];
    const float* bo = (const float*)d_in[10];

    float* out   = (float*)d_out;
    float* attnW = out + (size_t)B_ * S_ * D_;      // second tuple output

    char* ws = (char*)d_ws;
    const size_t WB = (size_t)D_ * D_ * sizeof(__bf16);       // 2 MB per weight
    const size_t TB = (size_t)B_ * S_ * D_ * sizeof(__bf16);  // 8 MB per tensor
    __bf16* Wq_b = (__bf16*)(ws + 0 * WB);
    __bf16* Wk_b = (__bf16*)(ws + 1 * WB);
    __bf16* Wv_b = (__bf16*)(ws + 2 * WB);
    __bf16* Wo_b = (__bf16*)(ws + 3 * WB);
    __bf16* Qb   = (__bf16*)(ws + 4 * WB);
    __bf16* Kb   = (__bf16*)(ws + 4 * WB + 1 * TB);
    __bf16* Vtb  = (__bf16*)(ws + 4 * WB + 2 * TB);
    __bf16* Cb   = (__bf16*)(ws + 4 * WB + 3 * TB);           // total ~40 MB

    const int nW = D_ * D_;
    cvt_kernel<<<nW / 256, 256, 0, stream>>>(Wq, Wq_b, nW);
    cvt_kernel<<<nW / 256, 256, 0, stream>>>(Wk, Wk_b, nW);
    cvt_kernel<<<nW / 256, 256, 0, stream>>>(Wv, Wv_b, nW);
    cvt_kernel<<<nW / 256, 256, 0, stream>>>(Wo, Wo_b, nW);

    proj_gemm_kernel<<<512, 256, 0, stream>>>(q, Wq_b, bq, Qb, 0);
    proj_gemm_kernel<<<512, 256, 0, stream>>>(k, Wk_b, bk, Kb, 0);
    proj_gemm_kernel<<<512, 256, 0, stream>>>(v, Wv_b, bv, Vtb, 1);

    const size_t smemB = (size_t)(16 * STR + 256) * sizeof(float);
    attn_kernel<<<B_ * H_ * (S_ / 16), 256, smemB, stream>>>(Qb, Kb, Vtb, attnW, Cb);

    out_gemm_kernel<<<512, 256, 0, stream>>>(Cb, Wo_b, bo, out);
}
